// EquivariantBlock_77738908057954
// MI455X (gfx1250) — matrix operand order repaired and checked by reading, not compiled, not run on hardware
//
#include <hip/hip_runtime.h>
#include <hip/hip_bf16.h>

// ---------------------------------------------------------------------------
// EGNN EquivariantBlock for gfx1250 (MI455X).
// N=512 nodes, E=512*511 edges, HID=128.
// Strategy: factor the 257->128 edge layer-1 through per-node projections
// (P = h@W_top + b, Q = h@W_bot), then per-edge work is only the two
// 128x128 layer-2 GEMMs, done with v_wmma_f32_16x16x32_f16.
// ---------------------------------------------------------------------------

typedef __attribute__((ext_vector_type(16))) _Float16 v16h;
typedef __attribute__((ext_vector_type(8)))  _Float16 v8h;
typedef __attribute__((ext_vector_type(8)))  float    v8f;

#define NPART 512
#define NEDGE 511
#define HID   128

union V16U { v16h v; struct { v8h lo, hi; } p; };

__device__ __forceinline__ float siluf(float z) { return z * (1.0f / (1.0f + __expf(-z))); }
__device__ __forceinline__ float sigmf(float z) { return 1.0f / (1.0f + __expf(-z)); }
// Compiler-level fence: CDNA5 LDS ops from one wave are executed in order by
// HW, we only need to stop the compiler from reordering shared accesses.
__device__ __forceinline__ void wave_fence() { asm volatile("" ::: "memory"); }

// ---------------------------------------------------------------------------
// Kernel 1: C[512x128] = A[512x128] @ W[128x128] (+ bias). One wave per 16x16
// output tile, K swept with 4x v_wmma_f32_16x16x32_f16.
// ---------------------------------------------------------------------------
__global__ void __launch_bounds__(32)
egnn_proj_kernel(const float* __restrict__ A, const float* __restrict__ W,
                 const float* __restrict__ bias, float* __restrict__ C)
{
    const int lane = threadIdx.x & 31;
    const int nt   = blockIdx.x;             // 0..7  (col tile)
    const int mt   = blockIdx.y;             // 0..31 (row tile)
    const int row  = mt * 16 + (lane & 15);
    const int col  = nt * 16 + (lane & 15);
    const int aoff = (lane < 16) ? 0 : 8;
    const int boff = (lane < 16) ? 0 : 16;

    v8f acc = {0.f, 0.f, 0.f, 0.f, 0.f, 0.f, 0.f, 0.f};
#pragma unroll
    for (int kc = 0; kc < 4; ++kc) {
        V16U a, b;
#pragma unroll
        for (int f = 0; f < 8; ++f) {
            a.v[f]     = (_Float16)A[row * HID + kc * 32 + boff + f];
            a.v[f + 8] = (_Float16)A[row * HID + kc * 32 + boff + 8 + f];
        }
#pragma unroll
        for (int f = 0; f < 16; ++f)
            b.v[f] = (_Float16)W[(kc * 32 + boff + f) * HID + col];
        acc = __builtin_amdgcn_wmma_f32_16x16x32_f16(false, a.v, false, b.v,
                                                     (short)0, acc, false, false);
    }
    const float bb = bias ? bias[col] : 0.f;
#pragma unroll
    for (int r = 0; r < 8; ++r) {
        const int ro = mt * 16 + r + ((lane < 16) ? 0 : 8);
        C[ro * HID + col] = acc[r] + bb;
    }
}

// ---------------------------------------------------------------------------
// Kernel 2: one block (4 waves) per node p. Each wave owns 8 tiles of 16
// edges. Phase A = coord branch (x update), phase B = edge/message branch
// (agg), phase C = node MLP + outputs.
// ---------------------------------------------------------------------------
__global__ void __launch_bounds__(128)
egnn_edge_kernel(const float* __restrict__ x,   const float* __restrict__ h,
                 const float* __restrict__ cw1, const float* __restrict__ cw2,
                 const float* __restrict__ cb2, const float* __restrict__ cw3,
                 const float* __restrict__ cb3, const float* __restrict__ ew1,
                 const float* __restrict__ ew2, const float* __restrict__ eb2,
                 const float* __restrict__ nw1, const float* __restrict__ nb1,
                 const float* __restrict__ nw2, const float* __restrict__ nb2,
                 const float* __restrict__ aw,  const int*   __restrict__ idx,
                 const float* __restrict__ Pc,  const float* __restrict__ Qc,
                 const float* __restrict__ Pe,  const float* __restrict__ Qe,
                 float* __restrict__ out_x,     float* __restrict__ out_h)
{
    // Swizzled layer-2 weights: frag(kc,nt) for lane L is 16 contiguous halfs.
    __shared__ __align__(32) _Float16 swz[HID * HID];        // 32 KB
    __shared__ __align__(32) _Float16 atile[4][16 * HID];    // 16 KB (A tile / m2e)
    __shared__ float red[4][16][16];                         // per-wave row-dot reduction
    __shared__ int   rm_j[4][16];
    __shared__ float rm_s[4][16];
    __shared__ float rm_w[4][16][4];                         // wx,wy,wz,valid
    __shared__ float rm_g[4][16];
    __shared__ float s_agg[HID];
    __shared__ float s_x[4];
    __shared__ float s_hh[HID];

    const int tid  = threadIdx.x;
    const int w    = tid >> 5;
    const int lane = tid & 31;
    const int p    = blockIdx.x;
    const int f0   = lane * 4;            // 4 features per lane (128/32)

    s_agg[tid] = 0.f;
    if (tid < 4) s_x[tid] = 0.f;

    const float xp0 = x[p * 3 + 0], xp1 = x[p * 3 + 1], xp2 = x[p * 3 + 2];
    _Float16* at = &atile[w][0];
    const int ao    = (lane < 16) ? 0 : 8;
    const int arow  = lane & 15;
    const int rbase = (lane < 16) ? 0 : 8;

    // ---------------- load cw2 fragment-swizzled ----------------
    for (int e = tid; e < HID * HID; e += 128) {
        const int k = e >> 7, n = e & 127;
        const int kc = k >> 5, koff = (k >> 4) & 1, f = k & 15;
        const int ntc = n >> 4, colp = n & 15;
        swz[(((kc * 8 + ntc) * 32 + koff * 16 + colp) << 4) + f] = (_Float16)cw2[e];
    }
    __syncthreads();

    // ======================= phase A: coordinate branch =====================
    const float4 pcv = *(const float4*)&Pc[p * HID + f0];
    const float4 wlc = *(const float4*)&cw1[256 * HID + f0];   // sqrd row of cw1
    const float  cb3v = cb3[0];
    float xa0 = 0.f, xa1 = 0.f, xa2 = 0.f;

    for (int it = 0; it < 8; ++it) {
        const int t = w * 8 + it;
        if (lane < 16) {                       // row metadata
            const int q = t * 16 + lane;
            int j = 0; float dx = 0.f, dy = 0.f, dz = 0.f, valid = 0.f;
            if (q < NEDGE) {
                j = idx[(p * NEDGE + q) * 2 + 1];
                dx = x[j * 3 + 0] - xp0; dy = x[j * 3 + 1] - xp1; dz = x[j * 3 + 2] - xp2;
                valid = 1.f;
            }
            const float s = dx * dx + dy * dy + dz * dz;
            const float inv = valid / (sqrtf(s) + 1.f);
            rm_j[w][lane] = j; rm_s[w][lane] = s;
            rm_w[w][lane][0] = dx * inv; rm_w[w][lane][1] = dy * inv;
            rm_w[w][lane][2] = dz * inv; rm_w[w][lane][3] = valid;
        }
        wave_fence();
        // A tile = silu(z1) in f16
#pragma unroll 4
        for (int r = 0; r < 16; ++r) {
            const int j = rm_j[w][r]; const float s = rm_s[w][r];
            const float4 qv = *(const float4*)&Qc[j * HID + f0];
            union { _Float16 hh[4]; uint2 u; } pk;
            pk.hh[0] = (_Float16)siluf(pcv.x + qv.x + s * wlc.x);
            pk.hh[1] = (_Float16)siluf(pcv.y + qv.y + s * wlc.y);
            pk.hh[2] = (_Float16)siluf(pcv.z + qv.z + s * wlc.z);
            pk.hh[3] = (_Float16)siluf(pcv.w + qv.w + s * wlc.w);
            *(uint2*)&at[r * HID + f0] = pk.u;
        }
        wave_fence();
        V16U af[4];
#pragma unroll
        for (int kc = 0; kc < 4; ++kc) {
            af[kc].p.lo = *(const v8h*)&at[arow * HID + kc * 32 + 2 * ao];
            af[kc].p.hi = *(const v8h*)&at[arow * HID + kc * 32 + 2 * ao + 8];
        }
        float cmp[8] = {0.f, 0.f, 0.f, 0.f, 0.f, 0.f, 0.f, 0.f};
#pragma unroll
        for (int nt = 0; nt < 8; ++nt) {
            v8f acc = {0.f, 0.f, 0.f, 0.f, 0.f, 0.f, 0.f, 0.f};
#pragma unroll
            for (int kc = 0; kc < 4; ++kc) {
                const v16h b = *(const v16h*)&swz[((kc * 8 + nt) * 32 + lane) << 4];
                acc = __builtin_amdgcn_wmma_f32_16x16x32_f16(false, af[kc].v, false, b,
                                                             (short)0, acc, false, false);
            }
            const int col = nt * 16 + (lane & 15);
            const float bb = cb2[col], w3 = cw3[col];
#pragma unroll
            for (int r = 0; r < 8; ++r) cmp[r] += siluf(acc[r] + bb) * w3;
        }
#pragma unroll
        for (int r = 0; r < 8; ++r) red[w][rbase + r][lane & 15] = cmp[r];
        wave_fence();
        if (lane < 16) {                       // finish cm, accumulate x update
            float cm = cb3v;
#pragma unroll
            for (int c = 0; c < 16; ++c) cm += red[w][lane][c];
            xa0 += rm_w[w][lane][0] * cm;
            xa1 += rm_w[w][lane][1] * cm;
            xa2 += rm_w[w][lane][2] * cm;
        }
        wave_fence();
    }
    if (lane < 16) {
        atomicAdd(&s_x[0], xa0); atomicAdd(&s_x[1], xa1); atomicAdd(&s_x[2], xa2);
    }
    __syncthreads();

    // ---------------- swap in ew2 fragment-swizzled ----------------
    for (int e = tid; e < HID * HID; e += 128) {
        const int k = e >> 7, n = e & 127;
        const int kc = k >> 5, koff = (k >> 4) & 1, f = k & 15;
        const int ntc = n >> 4, colp = n & 15;
        swz[(((kc * 8 + ntc) * 32 + koff * 16 + colp) << 4) + f] = (_Float16)ew2[e];
    }
    __syncthreads();

    // ======================= phase B: edge / message branch =================
    const float4 pev = *(const float4*)&Pe[p * HID + f0];
    const float4 wle = *(const float4*)&ew1[256 * HID + f0];

    for (int it = 0; it < 8; ++it) {
        const int t = w * 8 + it;
        if (lane < 16) {
            const int q = t * 16 + lane;
            int j = 0; float dx = 0.f, dy = 0.f, dz = 0.f, valid = 0.f;
            if (q < NEDGE) {
                j = idx[(p * NEDGE + q) * 2 + 1];
                dx = x[j * 3 + 0] - xp0; dy = x[j * 3 + 1] - xp1; dz = x[j * 3 + 2] - xp2;
                valid = 1.f;
            }
            rm_j[w][lane] = j;
            rm_s[w][lane] = dx * dx + dy * dy + dz * dz;
            rm_w[w][lane][3] = valid;
        }
        wave_fence();
#pragma unroll 4
        for (int r = 0; r < 16; ++r) {
            const int j = rm_j[w][r]; const float s = rm_s[w][r];
            const float4 qv = *(const float4*)&Qe[j * HID + f0];
            union { _Float16 hh[4]; uint2 u; } pk;
            pk.hh[0] = (_Float16)siluf(pev.x + qv.x + s * wle.x);
            pk.hh[1] = (_Float16)siluf(pev.y + qv.y + s * wle.y);
            pk.hh[2] = (_Float16)siluf(pev.z + qv.z + s * wle.z);
            pk.hh[3] = (_Float16)siluf(pev.w + qv.w + s * wle.w);
            *(uint2*)&at[r * HID + f0] = pk.u;
        }
        wave_fence();
        V16U af[4];
#pragma unroll
        for (int kc = 0; kc < 4; ++kc) {
            af[kc].p.lo = *(const v8h*)&at[arow * HID + kc * 32 + 2 * ao];
            af[kc].p.hi = *(const v8h*)&at[arow * HID + kc * 32 + 2 * ao + 8];
        }
        float gp[8] = {0.f, 0.f, 0.f, 0.f, 0.f, 0.f, 0.f, 0.f};
#pragma unroll
        for (int nt = 0; nt < 8; ++nt) {
            v8f acc = {0.f, 0.f, 0.f, 0.f, 0.f, 0.f, 0.f, 0.f};
#pragma unroll
            for (int kc = 0; kc < 4; ++kc) {
                const v16h b = *(const v16h*)&swz[((kc * 8 + nt) * 32 + lane) << 4];
                acc = __builtin_amdgcn_wmma_f32_16x16x32_f16(false, af[kc].v, false, b,
                                                             (short)0, acc, false, false);
            }
            const int col = nt * 16 + (lane & 15);
            const float bb = eb2[col], av = aw[col];
#pragma unroll
            for (int r = 0; r < 8; ++r) {
                const float m = siluf(acc[r] + bb);
                gp[r] += m * av;
                at[(rbase + r) * HID + col] = (_Float16)m;   // stash m2e for gating
            }
        }
#pragma unroll
        for (int r = 0; r < 8; ++r) red[w][rbase + r][lane & 15] = gp[r];
        wave_fence();
        if (lane < 16) {
            float g = 0.f;
#pragma unroll
            for (int c = 0; c < 16; ++c) g += red[w][lane][c];
            rm_g[w][lane] = sigmf(g) * rm_w[w][lane][3];     // gate * validity
        }
        wave_fence();
        // gated column sums into block-shared agg
        float a0 = 0.f, a1 = 0.f, a2 = 0.f, a3 = 0.f;
#pragma unroll 4
        for (int r = 0; r < 16; ++r) {
            const float g = rm_g[w][r];
            union { uint2 u; _Float16 hh[4]; } pk;
            pk.u = *(const uint2*)&at[r * HID + f0];
            a0 += (float)pk.hh[0] * g; a1 += (float)pk.hh[1] * g;
            a2 += (float)pk.hh[2] * g; a3 += (float)pk.hh[3] * g;
        }
        atomicAdd(&s_agg[f0 + 0], a0); atomicAdd(&s_agg[f0 + 1], a1);
        atomicAdd(&s_agg[f0 + 2], a2); atomicAdd(&s_agg[f0 + 3], a3);
        wave_fence();
    }
    __syncthreads();

    // ======================= phase C: node MLP + outputs ====================
    {
        const int f = tid;
        float acc = nb1[f];
        for (int k = 0; k < HID; ++k) acc += h[p * HID + k] * nw1[k * HID + f];
        for (int k = 0; k < HID; ++k) acc += s_agg[k] * nw1[(HID + k) * HID + f];
        s_hh[f] = siluf(acc);
    }
    __syncthreads();
    {
        const int f = tid;
        float o = nb2[f];
        for (int k = 0; k < HID; ++k) o += s_hh[k] * nw2[k * HID + f];
        out_h[p * HID + f] = o;
    }
    if (tid < 3) out_x[p * 3 + tid] = x[p * 3 + tid] + s_x[tid];
}

// ---------------------------------------------------------------------------
extern "C" void kernel_launch(void* const* d_in, const int* in_sizes, int n_in,
                              void* d_out, int out_size, void* d_ws, size_t ws_size,
                              hipStream_t stream)
{
    const float* x   = (const float*)d_in[0];
    const float* h   = (const float*)d_in[1];
    const float* cw1 = (const float*)d_in[2];
    const float* cb1 = (const float*)d_in[3];
    const float* cw2 = (const float*)d_in[4];
    const float* cb2 = (const float*)d_in[5];
    const float* cw3 = (const float*)d_in[6];
    const float* cb3 = (const float*)d_in[7];
    const float* ew1 = (const float*)d_in[8];
    const float* eb1 = (const float*)d_in[9];
    const float* ew2 = (const float*)d_in[10];
    const float* eb2 = (const float*)d_in[11];
    const float* nw1 = (const float*)d_in[12];
    const float* nb1 = (const float*)d_in[13];
    const float* nw2 = (const float*)d_in[14];
    const float* nb2 = (const float*)d_in[15];
    const float* aw  = (const float*)d_in[16];
    const int*   idx = (const int*)d_in[17];

    float* Pc = (float*)d_ws;                 // 4 x 512*128 f32 = 1 MB scratch
    float* Qc = Pc + NPART * HID;
    float* Pe = Qc + NPART * HID;
    float* Qe = Pe + NPART * HID;

    dim3 g1(8, 32), b1(32);
    egnn_proj_kernel<<<g1, b1, 0, stream>>>(h, cw1,             cb1,     Pc);
    egnn_proj_kernel<<<g1, b1, 0, stream>>>(h, cw1 + 128 * HID, nullptr, Qc);
    egnn_proj_kernel<<<g1, b1, 0, stream>>>(h, ew1,             eb1,     Pe);
    egnn_proj_kernel<<<g1, b1, 0, stream>>>(h, ew1 + 128 * HID, nullptr, Qe);

    float* out_x = (float*)d_out;
    float* out_h = out_x + NPART * 3;
    egnn_edge_kernel<<<NPART, 128, 0, stream>>>(x, h, cw1, cw2, cb2, cw3, cb3,
                                                ew1, ew2, eb2, nw1, nb1, nw2, nb2,
                                                aw, idx, Pc, Qc, Pe, Qe, out_x, out_h);
}
